// ClassLinearWithLORA_6992206758144
// MI455X (gfx1250) — matrix operand order repaired and checked by reading, not compiled, not run on hardware
//
#include <hip/hip_runtime.h>
#include <hip/hip_bf16.h>

// Problem constants (fixed by reference setup_inputs)
#define M_TOT   8192      // B*S = 4*2048
#define K_TOT   1024      // D_IN
#define N_TOT   4096      // D_OUT
#define R_LORA  16
#define C_CLS   8

typedef __attribute__((ext_vector_type(16))) __bf16 v16bf;
typedef __attribute__((ext_vector_type(8)))  __bf16 v8bf;
typedef __attribute__((ext_vector_type(2)))  __bf16 v2bf;
typedef __attribute__((ext_vector_type(8)))  float  v8f;
typedef __attribute__((ext_vector_type(4)))  float  v4f;

union BFrag {
    v16bf v;
    v2bf  p[8];
    unsigned short s[16];
};

// scalar f32 -> bf16 RNE (epilogue-only / fallback)
__device__ __forceinline__ unsigned short f32_bf16(float f) {
    union { float f; unsigned u; } cv; cv.f = f;
    unsigned u = cv.u;
    u += 0x7FFFu + ((u >> 16) & 1u);
    return (unsigned short)(u >> 16);
}

// Pack 8 floats (two float4) into 8 bf16 using the hardware packed converter
// (v_cvt_pk_bf16_f32: 2 elements per VALU op, RNE) when available.
__device__ __forceinline__ void pack8(BFrag& f, int half, float4 a, float4 b) {
#if __has_builtin(__builtin_amdgcn_cvt_pk_bf16_f32)
    f.p[half * 4 + 0] = __builtin_amdgcn_cvt_pk_bf16_f32(a.x, a.y);
    f.p[half * 4 + 1] = __builtin_amdgcn_cvt_pk_bf16_f32(a.z, a.w);
    f.p[half * 4 + 2] = __builtin_amdgcn_cvt_pk_bf16_f32(b.x, b.y);
    f.p[half * 4 + 3] = __builtin_amdgcn_cvt_pk_bf16_f32(b.z, b.w);
#elif defined(__clang__)
    v4f va = {a.x, a.y, a.z, a.w};
    v4f vb = {b.x, b.y, b.z, b.w};
    typedef __attribute__((ext_vector_type(4))) __bf16 v4bf;
    *(v4bf*)&f.s[half * 8 + 0] = __builtin_convertvector(va, v4bf);
    *(v4bf*)&f.s[half * 8 + 4] = __builtin_convertvector(vb, v4bf);
#else
    unsigned short* s = &f.s[half * 8];
    s[0] = f32_bf16(a.x); s[1] = f32_bf16(a.y);
    s[2] = f32_bf16(a.z); s[3] = f32_bf16(a.w);
    s[4] = f32_bf16(b.x); s[5] = f32_bf16(b.y);
    s[6] = f32_bf16(b.z); s[7] = f32_bf16(b.w);
#endif
}

// ---------------------------------------------------------------------------
// Kernel 1: t[m, r] = sum_k x[m, k] * A7[k, r]     (skinny GEMM, R=16)
// ---------------------------------------------------------------------------
__global__ __launch_bounds__(256)
void lora_t_kernel(const float* __restrict__ x,
                   const float* __restrict__ A7,
                   float* __restrict__ t) {
    const int r = threadIdx.x & 15;
    const int m = blockIdx.x * 16 + (threadIdx.x >> 4);
    float acc = 0.f;
    const float* xr = x + (size_t)m * K_TOT;
    #pragma unroll 8
    for (int k = 0; k < K_TOT; ++k) {
        acc = fmaf(xr[k], A7[k * R_LORA + r], acc);
    }
    t[m * R_LORA + r] = acc;
}

// ---------------------------------------------------------------------------
// Kernel 2: out = x @ W^T + b + (alpha*gate7) * (t @ B7)   via bf16 WMMA
// Workgroup: 256 threads = 8 waves (wave32). WG tile 128(M) x 128(N).
// Wave tile: 32(M) x 64(N) = 2 x 4 accumulators of 16x16.
// ---------------------------------------------------------------------------
__global__ __launch_bounds__(256)
void lora_gemm_wmma(const float* __restrict__ x,      // [M_TOT, K_TOT]
                    const float* __restrict__ W,      // [N_TOT, K_TOT]
                    const float* __restrict__ bias,   // [N_TOT]
                    const float* __restrict__ B7,     // [R_LORA, N_TOT]
                    const float* __restrict__ g7,     // [M_TOT]
                    const float* __restrict__ t,      // [M_TOT, R_LORA]
                    const int*   __restrict__ alpha_p,
                    float* __restrict__ out) {        // [M_TOT, N_TOT]
    const int lane = threadIdx.x & 31;
    const int wave = threadIdx.x >> 5;
    const int wm   = wave & 3;            // 4 M-strips of 32
    const int wn   = wave >> 2;           // 2 N-strips of 64
    const int rm   = blockIdx.y * 128 + wm * 32;
    const int cn   = blockIdx.x * 128 + wn * 64;
    const int l16  = lane & 15;
    const int hi   = lane >> 4;           // which 16-lane half

    v8f acc[2][4] = {};

    const float* xbase0 = x + (size_t)(rm + l16) * K_TOT;
    const float* xbase1 = x + (size_t)(rm + 16 + l16) * K_TOT;

    for (int k0 = 0; k0 < K_TOT; k0 += 32) {
        // ---- A fragments: 2 x (16x32 bf16). ISA layout:
        // lanes 0-15 : row = l16, K = {k0+0..7, k0+16..23}
        // lanes 16-31: row = l16, K = {k0+8..15, k0+24..31}
        BFrag a[2];
        #pragma unroll
        for (int mi = 0; mi < 2; ++mi) {
            const float* xr = (mi ? xbase1 : xbase0) + k0 + hi * 8;
            float4 p0 = *(const float4*)(xr);
            float4 p1 = *(const float4*)(xr + 4);
            float4 p2 = *(const float4*)(xr + 16);
            float4 p3 = *(const float4*)(xr + 20);
            pack8(a[mi], 0, p0, p1);
            pack8(a[mi], 1, p2, p3);
            __builtin_prefetch(xr + 32, 0, 0);        // next K tile -> global_prefetch_b8
        }
        // ---- B fragments: 4 x (32x16 bf16). ISA layout:
        // lanes 0-15 : col = l16, K = {k0+0..7, k0+16..23}
        // lanes 16-31: col = l16, K = {k0+8..15, k0+24..31}
        BFrag bf[4];
        #pragma unroll
        for (int ni = 0; ni < 4; ++ni) {
            const float* wr = W + (size_t)(cn + ni * 16 + l16) * K_TOT + k0 + hi * 8;
            float4 q0 = *(const float4*)(wr);
            float4 q1 = *(const float4*)(wr + 4);
            float4 q2 = *(const float4*)(wr + 16);
            float4 q3 = *(const float4*)(wr + 20);
            pack8(bf[ni], 0, q0, q1);
            pack8(bf[ni], 1, q2, q3);
            __builtin_prefetch(wr + 32, 0, 0);
        }
        // ---- 8 WMMAs per K-step per wave
        #pragma unroll
        for (int mi = 0; mi < 2; ++mi)
            #pragma unroll
            for (int ni = 0; ni < 4; ++ni)
                acc[mi][ni] = __builtin_amdgcn_wmma_f32_16x16x32_bf16(
                    false, a[mi].v, false, bf[ni].v,
                    (short)0, acc[mi][ni], false, false);
    }

    // ------------------- LoRA epilogue as one zero-padded WMMA -------------
    // A_lora = (alpha * gate7[row]) * t[row, 0..15], K padded 16->32 w/ zeros.
    const float alpha = (float)(*alpha_p);
    BFrag al[2];
    #pragma unroll
    for (int mi = 0; mi < 2; ++mi) {
        const int row = rm + mi * 16 + l16;
        const float g = alpha * g7[row];
        const float* tr = t + (size_t)row * R_LORA + hi * 8;   // K 0..7 or 8..15
        float4 t0 = *(const float4*)(tr);
        float4 t1 = *(const float4*)(tr + 4);
        t0.x *= g; t0.y *= g; t0.z *= g; t0.w *= g;
        t1.x *= g; t1.y *= g; t1.z *= g; t1.w *= g;
        pack8(al[mi], 0, t0, t1);
        #pragma unroll
        for (int e = 8; e < 16; ++e) al[mi].s[e] = 0;          // padded K region
    }
    BFrag bl[4];
    #pragma unroll
    for (int ni = 0; ni < 4; ++ni) {
        const int col = cn + ni * 16 + l16;
        {                                 // elems 0-7 carry K = hi*8..hi*8+7 (real R)
            #pragma unroll
            for (int e = 0; e < 8; ++e)
                bl[ni].s[e] = f32_bf16(B7[(size_t)(hi * 8 + e) * N_TOT + col]);
                                          // elems 8-15: K = 16..31 -> zero pad
            #pragma unroll
            for (int e = 8; e < 16; ++e) bl[ni].s[e] = 0;
        }
    }
    #pragma unroll
    for (int mi = 0; mi < 2; ++mi)
        #pragma unroll
        for (int ni = 0; ni < 4; ++ni)
            acc[mi][ni] = __builtin_amdgcn_wmma_f32_16x16x32_bf16(
                false, al[mi].v, false, bl[ni].v,
                (short)0, acc[mi][ni], false, false);

    // ------------------- bias add + store ---------------------------------
    // C/D layout: VGPR v, lanes 0-15 -> M = v, lanes 16-31 -> M = v + 8; N = l16
    #pragma unroll
    for (int ni = 0; ni < 4; ++ni) {
        const int col = cn + ni * 16 + l16;
        const float bv = bias[col];
        #pragma unroll
        for (int mi = 0; mi < 2; ++mi) {
            #pragma unroll
            for (int v = 0; v < 8; ++v) {
                const int row = rm + mi * 16 + v + hi * 8;
                out[(size_t)row * N_TOT + col] = acc[mi][ni][v] + bv;
            }
        }
    }
}

// ---------------------------------------------------------------------------
extern "C" void kernel_launch(void* const* d_in, const int* in_sizes, int n_in,
                              void* d_out, int out_size, void* d_ws, size_t ws_size,
                              hipStream_t stream) {
    const float* x     = (const float*)d_in[0];   // [4,2048,1024]
    const float* W     = (const float*)d_in[1];   // [4096,1024]
    const float* b     = (const float*)d_in[2];   // [4096]
    const float* A     = (const float*)d_in[3];   // [8,1024,16]
    const float* Bl    = (const float*)d_in[4];   // [8,16,4096]
    const float* gates = (const float*)d_in[5];   // [8,4,2048,1]
    const int*   alpha = (const int*)d_in[6];     // scalar 16
    float*       out   = (float*)d_out;

    const float* A7 = A     + (size_t)(C_CLS - 1) * K_TOT * R_LORA;
    const float* B7 = Bl    + (size_t)(C_CLS - 1) * R_LORA * N_TOT;
    const float* g7 = gates + (size_t)(C_CLS - 1) * M_TOT;

    float* t = (float*)d_ws;                      // [8192,16] = 512 KB scratch

    lora_t_kernel<<<M_TOT / 16, 256, 0, stream>>>(x, A7, t);

    dim3 grid(N_TOT / 128, M_TOT / 128);          // 32 x 64 workgroups
    lora_gemm_wmma<<<grid, 256, 0, stream>>>(x, W, b, B7, g7, t, alpha, out);
}
